// ContentMultiheadAttention_3925600109004
// MI455X (gfx1250) — matrix operand order repaired and checked by reading, not compiled, not run on hardware
//
#include <hip/hip_runtime.h>
#include <hip/hip_bf16.h>
#include <cstddef>
#include <cstdint>

// Problem constants (match reference)
constexpr int LQ  = 1024;  // sequence length
constexpr int BB  = 4;     // batch
constexpr int EE  = 1024;  // embed dim
constexpr int HH  = 16;    // heads
constexpr int HDD = 64;    // head dim
constexpr int MROWS = LQ * BB;  // 4096 rows for [L*B, E] GEMMs

typedef __attribute__((ext_vector_type(16))) _Float16 v16h;
typedef __attribute__((ext_vector_type(8)))  _Float16 v8h;
typedef __attribute__((ext_vector_type(8)))  float    v8f;

#if __has_builtin(__builtin_amdgcn_tensor_load_to_lds) && __has_builtin(__builtin_amdgcn_s_wait_tensorcnt)
#define ATHENA_TDM 1
#else
#define ATHENA_TDM 0
#endif

#if ATHENA_TDM
typedef unsigned int u32x4 __attribute__((ext_vector_type(4)));
typedef int          i32x4 __attribute__((ext_vector_type(4)));
typedef int          i32x8 __attribute__((ext_vector_type(8)));

// Issue one TDM transfer: 64 rows x 32 halfs (64B/row) from global (row stride
// K=1024 halfs) into LDS, with 16B padding appended after every 64B row so the
// LDS row stride is 80B (bank-conflict-free for the 16-lane B-operand reads).
__device__ __forceinline__ void tdm_load_tile(unsigned lds_addr, const _Float16* gsrc) {
    const unsigned long long ga = (unsigned long long)(uintptr_t)gsrc;
    u32x4 g0;
    g0[0] = 1u;                                            // count=1, user descriptor
    g0[1] = lds_addr;                                      // LDS byte address
    g0[2] = (unsigned)ga;                                  // global_addr[31:0]
    g0[3] = (unsigned)((ga >> 32) & 0x1FFFFFFu) | (2u << 30); // addr[56:32] | type=2
    i32x8 g1;
    g1[0] = (int)((1u << 16)      // data_size = 1 -> 2 bytes
                | (1u << 20)      // pad_enable
                | (3u << 22)      // pad_interval code 3 -> every 16 DWORDs (64B)
                | (3u << 25));    // pad_amount  code 3 -> 4 DWORDs (16B)
    g1[1] = (int)((1024u & 0xFFFFu) << 16);  // tensor_dim0[15:0] = 1024 (bits 63:48)
    g1[2] = (int)(64u << 16);                // tensor_dim0 hi = 0; tensor_dim1 = 64
    g1[3] = (int)(32u << 16);                // tensor_dim1 hi = 0; tile_dim0 = 32
    g1[4] = 64;                              // tile_dim1 = 64; tile_dim2 = 0
    g1[5] = 1024;                            // tensor_dim0_stride lo = 1024
    g1[6] = 0;                               // stride hi = 0; tensor_dim1_stride = 0
    g1[7] = 0;
    i32x4 z4 = {0, 0, 0, 0};
#if __clang_major__ >= 23
    i32x8 z8 = {0, 0, 0, 0, 0, 0, 0, 0};
    __builtin_amdgcn_tensor_load_to_lds(g0, g1, z4, z4, z8, 0);
#else
    __builtin_amdgcn_tensor_load_to_lds(g0, g1, z4, z4, 0);
#endif
}
#endif  // ATHENA_TDM

// Concatenate two contiguous 8-half (16B) chunks into a v16h WMMA operand.
__device__ __forceinline__ v16h ld16h(const _Float16* p0, const _Float16* p1) {
    v8h x = *(const v8h*)p0;
    v8h y = *(const v8h*)p1;
    v16h r;
#pragma unroll
    for (int i = 0; i < 8; ++i) { r[i] = x[i]; r[8 + i] = y[i]; }
    return r;
}

// 16 contiguous halfs (32B, aligned) as one operand.
__device__ __forceinline__ v16h ld16h_c(const _Float16* p) {
    return *(const v16h*)p;
}

// Load 8+8 floats (e.g. from LDS) and convert to f16 operand (P matrix path).
__device__ __forceinline__ v16h cvt16(const float* p0, const float* p1) {
    float t[16];
    *(float4*)&t[0]  = *(const float4*)(p0);
    *(float4*)&t[4]  = *(const float4*)(p0 + 4);
    *(float4*)&t[8]  = *(const float4*)(p1);
    *(float4*)&t[12] = *(const float4*)(p1 + 4);
    v16h a;
#pragma unroll
    for (int i = 0; i < 16; ++i) a[i] = (_Float16)t[i];
    return a;
}

// -----------------------------------------------------------------------------
// Elementwise f32 -> f16 conversion (vectorized x8).
// -----------------------------------------------------------------------------
__global__ __launch_bounds__(256)
void cvt_f32_to_f16(const float* __restrict__ in, _Float16* __restrict__ out, int n)
{
    const int i = (blockIdx.x * 256 + threadIdx.x) * 8;
    if (i + 8 <= n) {
        float4 a = *(const float4*)(in + i);
        float4 b = *(const float4*)(in + i + 4);
        v8h o;
        o[0] = (_Float16)a.x; o[1] = (_Float16)a.y; o[2] = (_Float16)a.z; o[3] = (_Float16)a.w;
        o[4] = (_Float16)b.x; o[5] = (_Float16)b.y; o[6] = (_Float16)b.z; o[7] = (_Float16)b.w;
        *(v8h*)(out + i) = o;
    }
}

// -----------------------------------------------------------------------------
// Y[M,N] = X[M,K] @ W[N,K]^T + bias[N].  X,W are f16; accumulate f32.
// MODE 0: store f16 row-major [M,N]
// MODE 1: store f16 V-transposed: Vt[(bb*EE + n)*LQ + l], m = l*BB + bb
// MODE 2: store f32 row-major [M,N]  (final output)
// 256 threads = 8 waves; each wave: 16(M) x 64(N) via 4 WMMAs per K-step.
// Block tile 128x64; the shared 64x32 B tile is staged into LDS once per block
// per K-step by the Tensor Data Mover (double-buffered), TENSORcnt-tracked.
// grid = (N/64, M/128). K must be 1024 (descriptor hardcodes strides).
// -----------------------------------------------------------------------------
template<int MODE>
__global__ __launch_bounds__(256)
void gemm_f16(const _Float16* __restrict__ X, const _Float16* __restrict__ W,
              const float* __restrict__ bias, void* __restrict__ Yv,
              int M, int N, int K)
{
    // Rows padded to 40 halfs (80B): conflict-free ds_load_b128 B-operand reads.
    __shared__ __align__(16) _Float16 sB[2][64][40];

    const int tid  = threadIdx.x;
    const int lane = tid & 31;
    const int wave = tid >> 5;
    const int m0 = blockIdx.y * 128 + wave * 16;
    const int n0 = blockIdx.x * 64;

    const int klo = (lane >> 4) * 8;    // unused; A and B both load at bko
    const int bko = (lane >> 4) * 16;   // B K-sub-offset {0,16}

    const _Float16* __restrict__ arow  = X + (size_t)(m0 + (lane & 15)) * K;
    const _Float16* __restrict__ wtile = W + (size_t)n0 * K;   // 64-row B panel

    v8f c0 = {}, c1 = {}, c2 = {}, c3 = {};

#if ATHENA_TDM
    if (wave == 0)
        tdm_load_tile((unsigned)(uintptr_t)&sB[0][0][0], wtile);   // kb = 0
#endif

    int buf = 0;
    for (int kb = 0; kb < K; kb += 32) {
        __syncthreads();   // everyone done reading the buffer we are about to refill
#if ATHENA_TDM
        if (wave == 0) {
            if (kb + 32 < K) {
                tdm_load_tile((unsigned)(uintptr_t)&sB[buf ^ 1][0][0], wtile + kb + 32);
                __builtin_amdgcn_s_wait_tensorcnt(1);   // current tile's DMA done
            } else {
                __builtin_amdgcn_s_wait_tensorcnt(0);
            }
        }
#else
        {   // cooperative staging fallback: 256 threads x 16B = 64 rows x 64B
            const int row = tid >> 2, qu = tid & 3;
            *(uint4*)&sB[buf][row][qu * 8] =
                *(const uint4*)(wtile + (size_t)row * K + kb + qu * 8);
        }
#endif
        __syncthreads();   // tile resident in LDS

        __builtin_prefetch(arow + kb + 128, 0, 1);
        v16h a  = ld16h_c(arow + kb + bko);
        v16h b0 = ld16h_c(&sB[buf][ 0 + (lane & 15)][bko]);
        v16h b1 = ld16h_c(&sB[buf][16 + (lane & 15)][bko]);
        v16h b2 = ld16h_c(&sB[buf][32 + (lane & 15)][bko]);
        v16h b3 = ld16h_c(&sB[buf][48 + (lane & 15)][bko]);
        c0 = __builtin_amdgcn_wmma_f32_16x16x32_f16(false, a, false, b0, (short)0, c0, false, false);
        c1 = __builtin_amdgcn_wmma_f32_16x16x32_f16(false, a, false, b1, (short)0, c1, false, false);
        c2 = __builtin_amdgcn_wmma_f32_16x16x32_f16(false, a, false, b2, (short)0, c2, false, false);
        c3 = __builtin_amdgcn_wmma_f32_16x16x32_f16(false, a, false, b3, (short)0, c3, false, false);
        buf ^= 1;
    }

    // C layout: VGPR r -> (M = mb+r, N = nsub + lane&15)
    const int mb = m0 + (lane >> 4) * 8;
    v8f acc[4] = {c0, c1, c2, c3};
#pragma unroll
    for (int j = 0; j < 4; ++j) {
        const int n = n0 + 16 * j + (lane & 15);
        const float bv = bias[n];
#pragma unroll
        for (int r = 0; r < 8; ++r) {
            const int m = mb + r;
            const float val = acc[j][r] + bv;
            if (MODE == 0) {
                ((_Float16*)Yv)[(size_t)m * N + n] = (_Float16)val;
            } else if (MODE == 1) {
                const int l  = m >> 2;     // m = l*BB + bb, BB = 4
                const int bb = m & 3;
                ((_Float16*)Yv)[((size_t)bb * EE + n) * LQ + l] = (_Float16)val;
            } else {
                ((float*)Yv)[(size_t)m * N + n] = val;
            }
        }
    }
}

// -----------------------------------------------------------------------------
// Attention core. One block per (q-tile of 16 rows, batch). Loops over heads.
// Q,K: f16 [L*B, E] (row = l*B+b, col = h*64+d). Vt: f16 [B, E, L] transposed.
// O: f16 [L*B, E]. avg: f32 [B, L, L] (block exclusively owns its slice).
// -----------------------------------------------------------------------------
__global__ __launch_bounds__(256)
void attn_core(const _Float16* __restrict__ Q, const _Float16* __restrict__ Km,
               const _Float16* __restrict__ Vt, const float* __restrict__ mask,
               _Float16* __restrict__ O, float* __restrict__ avg)
{
    __shared__ float sS[16][LQ];   // 64 KB score/prob tile (16 q rows x 1024 k)

    const int lane = threadIdx.x & 31;
    const int wave = threadIdx.x >> 5;
    const int qt = blockIdx.x;     // q tile index, 0..63
    const int b  = blockIdx.y;     // batch, 0..3
    const int q0 = qt * 16;

    const int klo = (lane >> 4) * 8;    // unused; A loads at bko too
    const int bko = (lane >> 4) * 16;   // B operand K-sub-offset

    for (int h = 0; h < HH; ++h) {
        // ---------------- S = (Q Kh^T) / sqrt(HD) ----------------
        {
            const int arow = q0 + (lane & 15);
            const _Float16* __restrict__ qb = Q + ((size_t)arow * BB + b) * EE + h * HDD;
            // Hoist the Q A-tiles: reused by all 8 k-tiles of this wave.
            const v16h a0 = ld16h_c(qb + bko);        // d = 0..31
            const v16h a1 = ld16h_c(qb + 32 + bko);   // d = 32..63
            for (int t = 0; t < 8; ++t) {
                const int kt  = wave * 8 + t;           // k tile 0..63
                const int brw = kt * 16 + (lane & 15);  // key sequence position
                const _Float16* __restrict__ kp = Km + ((size_t)brw * BB + b) * EE + h * HDD;
                v8f c = {};
                v16h b0 = ld16h_c(kp + bko);        // d = 0..31 slice
                v16h b1 = ld16h_c(kp + 32 + bko);   // d = 32..63 slice
                c = __builtin_amdgcn_wmma_f32_16x16x32_f16(false, a0, false, b0, (short)0, c, false, false);
                c = __builtin_amdgcn_wmma_f32_16x16x32_f16(false, a1, false, b1, (short)0, c, false, false);
                const int mb = (lane >> 4) * 8;
                const int n  = kt * 16 + (lane & 15);
#pragma unroll
                for (int r = 0; r < 8; ++r)
                    sS[mb + r][n] = c[r] * 0.125f;   // 1/sqrt(64)
            }
        }
        __syncthreads();

        // ---------------- row softmax (each wave: 2 rows) ----------------
#pragma unroll
        for (int rr = 0; rr < 2; ++rr) {
            const int r = wave * 2 + rr;
            const float* __restrict__ mrow = mask + ((size_t)b * LQ + (q0 + r)) * LQ;
            float mx = -3.402823466e38f;
            for (int k = lane; k < LQ; k += 32) {
                float s = sS[r][k] + mrow[k];
                sS[r][k] = s;
                mx = fmaxf(mx, s);
            }
#pragma unroll
            for (int off = 16; off > 0; off >>= 1)
                mx = fmaxf(mx, __shfl_xor(mx, off, 32));
            float sum = 0.f;
            for (int k = lane; k < LQ; k += 32) {
                float e = __expf(sS[r][k] - mx);
                sS[r][k] = e;
                sum += e;
            }
#pragma unroll
            for (int off = 16; off > 0; off >>= 1)
                sum += __shfl_xor(sum, off, 32);
            const float inv = 1.f / sum;
            for (int k = lane; k < LQ; k += 32)
                sS[r][k] *= inv;
        }
        __syncthreads();

        // ------------- averaged attention weights (block owns its slice) -------------
        {
            const float sc = 1.0f / (float)HH;
            for (int idx = threadIdx.x; idx < 16 * LQ; idx += 256) {
                const int r = idx >> 10;
                const int k = idx & (LQ - 1);
                float* p = avg + ((size_t)b * LQ + (q0 + r)) * LQ + k;
                const float v = sS[r][k] * sc;
                if (h == 0) *p = v; else *p += v;
            }
        }

        // ---------------- O tile = P @ V  (waves 0..3, one 16-col slab each) ----------------
        if (wave < 4) {
            const int nt = wave;
            const int d  = h * HDD + nt * 16 + (lane & 15);  // output column in E
            const int m  = lane & 15;                        // q row within tile
            const _Float16* __restrict__ vcol = Vt + ((size_t)b * EE + d) * LQ;
            v8f c = {};
            for (int kb = 0; kb < LQ; kb += 32) {
                const float* pa = &sS[m][kb + bko];
                v16h a  = cvt16(pa, pa + 8);            // probs from LDS -> f16
                v16h bm = ld16h_c(vcol + kb + bko);     // 16 consecutive positions
                c = __builtin_amdgcn_wmma_f32_16x16x32_f16(false, a, false, bm, (short)0, c, false, false);
            }
            const int mb = (lane >> 4) * 8;
#pragma unroll
            for (int r = 0; r < 8; ++r)
                O[((size_t)(q0 + mb + r) * BB + b) * EE + d] = (_Float16)c[r];
        }
        __syncthreads();   // sS reused next head
    }
}

// -----------------------------------------------------------------------------
// Host-side launch
// -----------------------------------------------------------------------------
extern "C" void kernel_launch(void* const* d_in, const int* in_sizes, int n_in,
                              void* d_out, int out_size, void* d_ws, size_t ws_size,
                              hipStream_t stream) {
    (void)in_sizes; (void)n_in; (void)out_size; (void)ws_size;

    const float* q     = (const float*)d_in[0];  // [L,B,E]
    const float* k     = (const float*)d_in[1];  // [L,B,E]
    const float* v     = (const float*)d_in[2];  // [L,B,E]
    const float* mask  = (const float*)d_in[3];  // [B,L,L]
    const float* w_in  = (const float*)d_in[4];  // [3E,E]
    const float* b_in  = (const float*)d_in[5];  // [3E]
    const float* w_out = (const float*)d_in[6];  // [E,E]
    const float* b_out = (const float*)d_in[7];  // [E]

    float* out      = (float*)d_out;
    float* attn_out = out;                                   // L*B*E f32
    float* avg      = out + (size_t)LQ * BB * EE;            // B*L*L f32

    const size_t plane = (size_t)MROWS * EE;                 // 4M elements
    _Float16* Xq = (_Float16*)d_ws;
    _Float16* Xk = Xq + plane;
    _Float16* Xv = Xk + plane;
    _Float16* Wi = Xv + plane;                               // 3M halfs
    _Float16* Wo = Wi + (size_t)3 * EE * EE;                 // 1M halfs
    _Float16* Qp = Wo + (size_t)EE * EE;
    _Float16* Kp = Qp + plane;
    _Float16* Vt = Kp + plane;                               // [B, E, L] transposed
    _Float16* Op = Vt + plane;

    dim3 blk(256);

    // f32 -> f16 staging
    cvt_f32_to_f16<<<dim3((unsigned)(plane / 2048)), blk, 0, stream>>>(q, Xq, (int)plane);
    cvt_f32_to_f16<<<dim3((unsigned)(plane / 2048)), blk, 0, stream>>>(k, Xk, (int)plane);
    cvt_f32_to_f16<<<dim3((unsigned)(plane / 2048)), blk, 0, stream>>>(v, Xv, (int)plane);
    cvt_f32_to_f16<<<dim3((unsigned)(3u * EE * EE / 2048)), blk, 0, stream>>>(w_in, Wi, 3 * EE * EE);
    cvt_f32_to_f16<<<dim3((unsigned)(EE * EE / 2048)), blk, 0, stream>>>(w_out, Wo, EE * EE);

    dim3 gg(EE / 64, MROWS / 128);   // (16, 32)

    // Packed in-projection
    gemm_f16<0><<<gg, blk, 0, stream>>>(Xq, Wi,                       b_in,          Qp, MROWS, EE, EE);
    gemm_f16<0><<<gg, blk, 0, stream>>>(Xk, Wi + (size_t)EE * EE,     b_in + EE,     Kp, MROWS, EE, EE);
    gemm_f16<1><<<gg, blk, 0, stream>>>(Xv, Wi + (size_t)2 * EE * EE, b_in + 2 * EE, Vt, MROWS, EE, EE);

    // Attention (scores, softmax, avg weights, P@V)
    attn_core<<<dim3(LQ / 16, BB), blk, 0, stream>>>(Qp, Kp, Vt, mask, Op, avg);

    // Out-projection -> f32 output
    gemm_f16<2><<<gg, blk, 0, stream>>>(Op, Wo, b_out, attn_out, MROWS, EE, EE);
}
